// Emulator_7413113553181
// MI455X (gfx1250) — matrix operand order repaired and checked by reading, compiled, not run on hardware
//
#include <hip/hip_runtime.h>
#include <hip/hip_bf16.h>

// ---- problem constants -----------------------------------------------------
#define B_   65536
#define T_   256
#define P_   4
#define L_   14
#define H_   32
#define IN_  18

#define WAVES 8     // waves per block
#define ROWS  16    // batch rows per wave (WMMA M)
#define CH    8     // timesteps of phys staged per async chunk

typedef __attribute__((ext_vector_type(16))) _Float16 v16h;
typedef __attribute__((ext_vector_type(8)))  _Float16 v8h;
typedef __attribute__((ext_vector_type(4)))  _Float16 v4h;
typedef __attribute__((ext_vector_type(8)))  float    v8f;

// ---- CDNA5 async copy (guarded: falls back to direct global loads) ---------
#ifndef __has_builtin
#define __has_builtin(x) 0
#endif
#if __has_builtin(__builtin_amdgcn_global_load_async_to_lds_b128)
#define USE_ASYNC 1
#else
#define USE_ASYNC 0
#endif

#if USE_ASYNC
// Builtin param 0 is a non-const AS(1) pointer to a gcc-vector int4 (per the
// compiler diagnostic); param 1 is the AS(3) LDS pointer.
typedef int v4i_ __attribute__((vector_size(16)));
#define GAS(p)   ((__attribute__((address_space(1))) v4i_*)(uintptr_t)(p))
#define LDSAS(p) ((__attribute__((address_space(3))) v4i_*)(unsigned)(uintptr_t)(p))
#endif

__device__ __forceinline__ float relu1(float x) {
#if __has_builtin(__builtin_amdgcn_fmed3f)
  return __builtin_amdgcn_fmed3f(x, 0.0f, __builtin_inff());  // 1x v_med3_f32
#else
  return fmaxf(x, 0.0f);
#endif
}

__device__ __forceinline__ v8f wmma_f16(v16h a, v16h b, v8f c) {
  // (neg_a, A, neg_b, B, c_mod, C, reuse_a, reuse_b)
  return __builtin_amdgcn_wmma_f32_16x16x32_f16(false, a, false, b, (short)0, c,
                                                false, false);
}

__device__ __forceinline__ v8f splat8(float x) {
  v8f v;
#pragma unroll
  for (int i = 0; i < 8; ++i) v[i] = x;
  return v;
}

// Load a 16x32 f16 A-fragment from a row-major [16 rows][32 K] LDS tile.
// 16-bit A layout: lane<16 -> K 0..7 | 16..23 ; lane>=16 -> K 8..15 | 24..31.
__device__ __forceinline__ v16h loadA(const _Float16* tile, int lane) {
  const int m    = lane & 15;
  const int koff = (lane & 16) ? 8 : 0;
  v8h lo = *(const v8h*)(tile + m * 32 + koff);        // ds_load_b128
  v8h hi = *(const v8h*)(tile + m * 32 + koff + 16);   // ds_load_b128
  return __builtin_shufflevector(lo, hi, 0, 1, 2, 3, 4, 5, 6, 7,
                                 8, 9, 10, 11, 12, 13, 14, 15);
}

// Store a relu'd f32 D tile (N-tile `nt`) into the row-major [16][32] f16 tile.
// D layout: lane -> col N=lane%16(+16*nt), rows r+8*(lane>=16).
__device__ __forceinline__ void storeD_relu(_Float16* tile, v8f d, int lane,
                                            int nt) {
  const int n  = (lane & 15) + 16 * nt;
  const int rb = (lane >> 4) << 3;
#pragma unroll
  for (int r = 0; r < 8; ++r)
    tile[(rb + r) * 32 + n] = (_Float16)relu1(d[r]);
}

__global__ __launch_bounds__(WAVES * 32)
void rnn_mlp_wmma(const float* __restrict__ phys,     // [B,T,P]
                  const float* __restrict__ latents,  // [B,L]
                  const float* __restrict__ W1,       // [IN,H]
                  const float* __restrict__ b1,       // [H]
                  const float* __restrict__ W2,       // [H,H]
                  const float* __restrict__ b2,       // [H]
                  const float* __restrict__ W3,       // [H,L]
                  const float* __restrict__ b3,       // [L]
                  float* __restrict__ out) {          // [B,T,L]
  __shared__ __align__(16) _Float16 sX[WAVES][ROWS * 32];  // x tile per wave
  __shared__ __align__(16) _Float16 sH[WAVES][ROWS * 32];  // h tile per wave
#if USE_ASYNC
  __shared__ __align__(16) float sP[WAVES][2][ROWS * CH * P_];  // phys stage
#endif

  const int lane = threadIdx.x & 31;
  const int w    = threadIdx.x >> 5;
  const int base = blockIdx.x * (WAVES * ROWS) + w * ROWS;  // first batch row
  const int n    = lane & 15;            // N column owned in B/C/D layouts
  const int rb   = (lane >> 4) << 3;     // 0 or 8: row base in C/D layout
  const int kb   = lane & 16;            // 0 or 16; kb/2 is the K base in B layout

  _Float16* xt = sX[w];
  _Float16* ht = sH[w];

  // ---- constant B-fragments (weights) and bias splats ----------------------
  v16h bw10, bw11, bw20, bw21, bw3;
#pragma unroll
  for (int h = 0; h < 16; ++h) {
    const int k = (kb >> 1) + (h & 7) + ((h & 8) << 1);
    bw10[h] = (k < IN_) ? (_Float16)W1[k * H_ + n]      : (_Float16)0.0f;
    bw11[h] = (k < IN_) ? (_Float16)W1[k * H_ + n + 16] : (_Float16)0.0f;
    bw20[h] = (_Float16)W2[k * H_ + n];
    bw21[h] = (_Float16)W2[k * H_ + n + 16];
    bw3[h]  = (n < L_) ? (_Float16)W3[k * L_ + n]       : (_Float16)0.0f;
  }
  const float bc10 = b1[n], bc11 = b1[n + 16];
  const float bc20 = b2[n], bc21 = b2[n + 16];
  const float bc3  = (n < L_) ? b3[n] : 0.0f;

  // ---- persistent f32 latent accumulator in C/D layout ---------------------
  v8f lat;
#pragma unroll
  for (int r = 0; r < 8; ++r)
    lat[r] = (n < L_) ? latents[(size_t)(base + rb + r) * L_ + n] : 0.0f;

  // zero the padded K columns 18..31 of the x tile once
  if (lane < 16) {
#pragma unroll
    for (int k = IN_; k < 32; ++k) xt[lane * 32 + k] = (_Float16)0.0f;
  }

#if USE_ASYNC
  // prime chunk 0 of phys into LDS (4 async b128 per wave)
  {
#pragma unroll
    for (int j = 0; j < (ROWS * CH * P_) / (32 * 4); ++j) {
      const int seg = lane + 32 * j;            // seg = m*CH + tr
      const int m = seg >> 3, tr = seg & (CH - 1);
      const float* g = &phys[((size_t)(base + m) * T_ + tr) * P_];
      __builtin_amdgcn_global_load_async_to_lds_b128(GAS(g),
                                                     LDSAS(&sP[w][0][seg * P_]),
                                                     0, 0);
    }
  }
  int buf = 0;
#endif

  for (int t = 0; t < T_; ++t) {
#if USE_ASYNC
    if ((t & (CH - 1)) == 0) {
      if (t + CH < T_) {
#pragma unroll
        for (int j = 0; j < (ROWS * CH * P_) / (32 * 4); ++j) {
          const int seg = lane + 32 * j;
          const int m = seg >> 3, tr = seg & (CH - 1);
          const float* g = &phys[((size_t)(base + m) * T_ + (t + CH + tr)) * P_];
          __builtin_amdgcn_global_load_async_to_lds_b128(
              GAS(g), LDSAS(&sP[w][buf ^ 1][seg * P_]), 0, 0);
        }
        asm volatile("s_wait_asynccnt 0x4" ::: "memory");  // current chunk done
      } else {
        asm volatile("s_wait_asynccnt 0x0" ::: "memory");
      }
    }
#endif

    // write latent (f32 C/D layout) into x tile as f16, K columns 4..17
    if (n < L_) {
#pragma unroll
      for (int r = 0; r < 8; ++r)
        xt[(rb + r) * 32 + P_ + n] = (_Float16)lat[r];
    }
    // write phys into x tile, K columns 0..3
    if (lane < 16) {
#if USE_ASYNC
      const float* pp = &sP[w][buf][(lane * CH + (t & (CH - 1))) * P_];
#else
      const float* pp = &phys[((size_t)(base + lane) * T_ + t) * P_];
#endif
      v4h ph;
#pragma unroll
      for (int p = 0; p < P_; ++p) ph[p] = (_Float16)pp[p];
      *(v4h*)(xt + lane * 32) = ph;  // ds_store_b64
    }

    // ---- layer 1: h1 = relu(x @ W1 + b1) ------------------------------
    v16h a  = loadA(xt, lane);
    v8f  d0 = wmma_f16(a, bw10, splat8(bc10));
    v8f  d1 = wmma_f16(a, bw11, splat8(bc11));
    storeD_relu(ht, d0, lane, 0);
    storeD_relu(ht, d1, lane, 1);

    // ---- layer 2: h2 = relu(h1 @ W2 + b2) -----------------------------
    a  = loadA(ht, lane);
    d0 = wmma_f16(a, bw20, splat8(bc20));
    d1 = wmma_f16(a, bw21, splat8(bc21));
    storeD_relu(ht, d0, lane, 0);
    storeD_relu(ht, d1, lane, 1);

    // ---- layer 3: lat += h2 @ W3 + b3 (residual via C operand) --------
    a   = loadA(ht, lane);
    lat = wmma_f16(a, bw3, lat);
#pragma unroll
    for (int r = 0; r < 8; ++r) lat[r] += bc3;

    // ---- emit out[b, t, :] as a non-temporal write-once stream --------
    if (n < L_) {
#pragma unroll
      for (int r = 0; r < 8; ++r)
        __builtin_nontemporal_store(
            lat[r], &out[((size_t)(base + rb + r) * T_ + t) * L_ + n]);
    }

#if USE_ASYNC
    if ((t & (CH - 1)) == (CH - 1)) buf ^= 1;
#endif
  }
}

extern "C" void kernel_launch(void* const* d_in, const int* in_sizes, int n_in,
                              void* d_out, int out_size, void* d_ws,
                              size_t ws_size, hipStream_t stream) {
  (void)in_sizes; (void)n_in; (void)out_size; (void)d_ws; (void)ws_size;
  const float* phys    = (const float*)d_in[0];
  const float* latents = (const float*)d_in[1];
  const float* W1      = (const float*)d_in[2];
  const float* b1      = (const float*)d_in[3];
  const float* W2      = (const float*)d_in[4];
  const float* b2      = (const float*)d_in[5];
  const float* W3      = (const float*)d_in[6];
  const float* b3      = (const float*)d_in[7];
  float* out           = (float*)d_out;

  const int rowsPerBlock = WAVES * ROWS;            // 128
  dim3 grid(B_ / rowsPerBlock);                     // 512 blocks
  dim3 block(WAVES * 32);                           // 256 threads (8 wave32)
  rnn_mlp_wmma<<<grid, block, 0, stream>>>(phys, latents, W1, b1, W2, b2, W3,
                                           b3, out);
}
